// PointAttention_29257317220738
// MI455X (gfx1250) — matrix operand order repaired and checked by reading, not compiled, not run on hardware
//
#include <hip/hip_runtime.h>

typedef __attribute__((ext_vector_type(16))) _Float16 v16h;
typedef __attribute__((ext_vector_type(8)))  _Float16 v8h;
typedef __attribute__((ext_vector_type(8)))  float    v8f;

namespace {
constexpr int kM = 32;        // points per voxel
constexpr int kD = 64;        // d_model
constexpr float kEps = 1e-5f;
constexpr float kScale = 0.125f;  // 1/sqrt(64)
}

__device__ __forceinline__ unsigned pack2(float a, float b) {
  union { _Float16 h[2]; unsigned u; } p;
  p.h[0] = (_Float16)a; p.h[1] = (_Float16)b;
  return p.u;
}

__global__ __launch_bounds__(256)
void point_attn_kernel(
    const float* __restrict__ pv_xyz,    // (B,32,3)
    const float* __restrict__ features,  // (B,32,64)
    const float* __restrict__ Wq, const float* __restrict__ bq,
    const float* __restrict__ Wk, const float* __restrict__ bk,
    const float* __restrict__ Wv, const float* __restrict__ bv,
    const float* __restrict__ pW1, const float* __restrict__ pb1,
    const float* __restrict__ plng, const float* __restrict__ plnb,
    const float* __restrict__ pW2, const float* __restrict__ pb2,
    const float* __restrict__ ln1g, const float* __restrict__ ln1b,
    const float* __restrict__ wW1, const float* __restrict__ wb1,
    const float* __restrict__ ln2g, const float* __restrict__ ln2b,
    const float* __restrict__ wW2, const float* __restrict__ wb2,
    float* __restrict__ out)             // (B,32,128)
{
  __shared__ float sx[kM * 3];
  __shared__ int   sEmpty[kM];
  __shared__ float sf[kM * kD];
  __shared__ float sq[kM * kD];
  __shared__ float sk[kM * kD];
  __shared__ float sv[kM * kD];
  // Weights transposed [n][pos] with K permuted: feature(pos) = (pos&3)*16 + (pos>>2)
  __shared__ __align__(16) _Float16 sW1T[kD * kD];
  __shared__ __align__(16) _Float16 sW2T[kD * kD];
  __shared__ float sb1v[kD], sb2v[kD];
  __shared__ float sg1[kD], sB1[kD], sg2[kD], sB2[kD];
  __shared__ float sPW2[3 * kD], sPb2[kD];
  __shared__ float sPsm[18];                            // pW1(9) pb1(3) plng(3) plnb(3)
  __shared__ __align__(16) _Float16 stage[8 * 16 * kD]; // per-wave A staging (K-permuted)

  const int b    = blockIdx.x;
  const int tid  = threadIdx.x;
  const int wave = tid >> 5;
  const int lane = tid & 31;
  const int half = lane >> 4;
  const int l16  = lane & 15;

  // ---------------- phase 1: global -> LDS ----------------
  for (int idx = tid; idx < kM * 3; idx += 256) sx[idx] = pv_xyz[b * kM * 3 + idx];
  for (int idx = tid; idx < kM * kD; idx += 256) {
    float f = features[b * kM * kD + idx];
    sf[idx] = f;
    out[b * kM * 128 + (idx >> 6) * 128 + 64 + (idx & 63)] = f;  // concat tail
  }
  for (int idx = tid; idx < kD * kD; idx += 256) {
    int n = idx >> 6, pos = idx & 63;
    int f = (pos & 3) * 16 + (pos >> 2);                 // K permutation
    sW1T[idx] = (_Float16)wW1[f * kD + n];
    sW2T[idx] = (_Float16)wW2[f * kD + n];
  }
  for (int idx = tid; idx < kD; idx += 256) {
    sb1v[idx] = wb1[idx];  sb2v[idx] = wb2[idx];
    sg1[idx]  = ln1g[idx]; sB1[idx]  = ln1b[idx];
    sg2[idx]  = ln2g[idx]; sB2[idx]  = ln2b[idx];
    sPb2[idx] = pb2[idx];
  }
  for (int idx = tid; idx < 3 * kD; idx += 256) sPW2[idx] = pW2[idx];
  if (tid < 9)       sPsm[tid] = pW1[tid];
  else if (tid < 12) sPsm[tid] = pb1[tid - 9];
  else if (tid < 15) sPsm[tid] = plng[tid - 12];
  else if (tid < 18) sPsm[tid] = plnb[tid - 15];
  __syncthreads();

  // ---------------- phase 2: empty flags, q, k, v ----------------
  for (int i = tid; i < kM; i += 256)
    sEmpty[i] = (sx[i * 3] == 0.f) & (sx[i * 3 + 1] == 0.f) & (sx[i * 3 + 2] == 0.f);
  for (int idx = tid; idx < kM * kD; idx += 256) {
    int i = idx >> 6, d = idx & 63;
    float x0 = sx[i * 3], x1 = sx[i * 3 + 1], x2 = sx[i * 3 + 2];
    sq[idx] = fmaf(x0, Wq[d], fmaf(x1, Wq[64 + d], fmaf(x2, Wq[128 + d], bq[d])));
    sk[idx] = fmaf(x0, Wk[d], fmaf(x1, Wk[64 + d], fmaf(x2, Wk[128 + d], bk[d])));
    float acc = bv[d];
    #pragma unroll 8
    for (int c = 0; c < kD; ++c) acc = fmaf(sf[i * kD + c], Wv[c * kD + d], acc);
    sv[idx] = acc;
  }
  __syncthreads();

  // ---------------- main loop: each wave owns 4 i's ----------------
  _Float16* st = &stage[wave * 16 * kD];

  #pragma clang loop unroll(disable)
  for (int ii = 0; ii < 4; ++ii) {
    const int i = wave + ii * 8;
    const float xi0 = sx[i * 3], xi1 = sx[i * 3 + 1], xi2 = sx[i * 3 + 2];
    const int iempty = sEmpty[i];

    float rm[4], rs[4], draw[4];
    #pragma unroll
    for (int t = 0; t < 4; ++t) { rm[t] = -3.4e38f; rs[t] = 0.f; draw[t] = 0.f; }

    #pragma clang loop unroll(disable)
    for (int bj = 0; bj < 2; ++bj) {
      // Launder weight bases so B-fragments reload from LDS each batch
      // (keeps VGPR count <=256 instead of caching 128 VGPRs of weights).
      unsigned lw = 0;
      asm volatile("" : "+v"(lw));
      const _Float16* W1p = &sW1T[lw];
      const _Float16* W2p = &sW2T[lw];

      const int j = bj * 16 + l16;  // this lane's row within the batch

      // ---- w0 row chunk (32 features per lane: d = half*32 + u) ----
      float d0 = fabsf(sx[j * 3]     - xi0);
      float d1 = fabsf(sx[j * 3 + 1] - xi1);
      float d2 = fabsf(sx[j * 3 + 2] - xi2);
      if (iempty | sEmpty[j]) { d0 = -1.f; d1 = -1.f; d2 = -1.f; }
      float p0 = fmaf(d0, sPsm[0], fmaf(d1, sPsm[3], fmaf(d2, sPsm[6], sPsm[9])));
      float p1 = fmaf(d0, sPsm[1], fmaf(d1, sPsm[4], fmaf(d2, sPsm[7], sPsm[10])));
      float p2 = fmaf(d0, sPsm[2], fmaf(d1, sPsm[5], fmaf(d2, sPsm[8], sPsm[11])));
      float mu3  = (p0 + p1 + p2) * (1.f / 3.f);
      float var3 = (p0 * p0 + p1 * p1 + p2 * p2) * (1.f / 3.f) - mu3 * mu3;
      float iv3  = rsqrtf(var3 + kEps);
      p0 = fmaxf(fmaf((p0 - mu3) * iv3, sPsm[12], sPsm[15]), 0.f);
      p1 = fmaxf(fmaf((p1 - mu3) * iv3, sPsm[13], sPsm[16]), 0.f);
      p2 = fmaxf(fmaf((p2 - mu3) * iv3, sPsm[14], sPsm[17]), 0.f);

      float w0[32];
      const int dbase = half * 32;
      float s1 = 0.f, s2 = 0.f;
      #pragma unroll
      for (int u = 0; u < 32; ++u) {
        int d = dbase + u;
        float pij = fmaf(p0, sPW2[d], fmaf(p1, sPW2[64 + d], fmaf(p2, sPW2[128 + d], sPb2[d])));
        float w = (sq[i * kD + d] - sk[j * kD + d] + pij) * kScale;
        w0[u] = w; s1 += w; s2 = fmaf(w, w, s2);
      }
      // LN over 64 (partner lane holds the other 32 features of the same row)
      s1 += __shfl_xor(s1, 16);
      s2 += __shfl_xor(s2, 16);
      float mu  = s1 * (1.f / 64.f);
      float inv = rsqrtf(s2 * (1.f / 64.f) - mu * mu + kEps);
      // Stage as f16 in K-permuted order: feature d -> pos (d&15)*4 + (d>>4).
      // Pair (u, u+16) lands at adjacent positions u*4 + 2*half + {0,1}.
      #pragma unroll
      for (int u = 0; u < 16; ++u) {
        int da = dbase + u, db2 = dbase + u + 16;
        float y0 = fmaxf(fmaf((w0[u]      - mu) * inv, sg1[da],  sB1[da]),  0.f);
        float y1 = fmaxf(fmaf((w0[u + 16] - mu) * inv, sg1[db2], sB1[db2]), 0.f);
        *(unsigned*)&st[l16 * kD + u * 4 + 2 * half] = pack2(y0, y1);
      }

      // ---- layer 1 GEMM: 16 rows x 64 @ W1 ----
      v8f c1[4] = {};
      #pragma unroll
      for (int ks = 0; ks < 2; ++ks) {
        const int a0 = ks * 32 + half * 8;               // ISA 16-bit A layout (positions)
        v8h alo = *(const v8h*)&st[l16 * kD + a0];
        v8h ahi = *(const v8h*)&st[l16 * kD + a0 + 16];
        v16h a;
        #pragma unroll
        for (int e = 0; e < 8; ++e) { a[e] = alo[e]; a[8 + e] = ahi[e]; }
        #pragma unroll
        for (int t = 0; t < 4; ++t) {
          const _Float16* wr = &W1p[(t * 16 + l16) * kD + ks * 32 + half * 8];
          v8h blo = *(const v8h*)wr;
          v8h bhi = *(const v8h*)(wr + 16);
          v16h bf;
          #pragma unroll
          for (int e = 0; e < 8; ++e) { bf[e] = blo[e]; bf[8 + e] = bhi[e]; }
          c1[t] = __builtin_amdgcn_wmma_f32_16x16x32_f16(
              false, a, false, bf, (short)0, c1[t], false, false);
        }
      }

      // ---- bias + LN2 + ReLU on C-layout, restage (K-permuted, b64 stores) ----
      #pragma unroll
      for (int t = 0; t < 4; ++t) {
        float bb = sb1v[t * 16 + l16];
        #pragma unroll
        for (int r = 0; r < 8; ++r) c1[t][r] += bb;
      }
      #pragma unroll
      for (int r = 0; r < 8; ++r) {
        float rsu = c1[0][r] + c1[1][r] + c1[2][r] + c1[3][r];
        float rsq = c1[0][r] * c1[0][r] + c1[1][r] * c1[1][r] +
                    c1[2][r] * c1[2][r] + c1[3][r] * c1[3][r];
        #pragma unroll
        for (int mk = 1; mk < 16; mk <<= 1) {
          rsu += __shfl_xor(rsu, mk);
          rsq += __shfl_xor(rsq, mk);
        }
        float mu2  = rsu * (1.f / 64.f);
        float inv2 = rsqrtf(rsq * (1.f / 64.f) - mu2 * mu2 + kEps);
        int row = r + half * 8;
        float y[4];
        #pragma unroll
        for (int t = 0; t < 4; ++t) {
          int n = t * 16 + l16;                     // semantic feature n -> pos l16*4 + t
          y[t] = fmaxf(fmaf((c1[t][r] - mu2) * inv2, sg2[n], sB2[n]), 0.f);
        }
        uint2 pk;
        pk.x = pack2(y[0], y[1]);
        pk.y = pack2(y[2], y[3]);
        *(uint2*)&st[row * kD + l16 * 4] = pk;
      }

      // ---- layer 2 GEMM ----
      v8f c2[4] = {};
      #pragma unroll
      for (int ks = 0; ks < 2; ++ks) {
        const int a0 = ks * 32 + half * 8;
        v8h alo = *(const v8h*)&st[l16 * kD + a0];
        v8h ahi = *(const v8h*)&st[l16 * kD + a0 + 16];
        v16h a;
        #pragma unroll
        for (int e = 0; e < 8; ++e) { a[e] = alo[e]; a[8 + e] = ahi[e]; }
        #pragma unroll
        for (int t = 0; t < 4; ++t) {
          const _Float16* wr = &W2p[(t * 16 + l16) * kD + ks * 32 + half * 8];
          v8h blo = *(const v8h*)wr;
          v8h bhi = *(const v8h*)(wr + 16);
          v16h bf;
          #pragma unroll
          for (int e = 0; e < 8; ++e) { bf[e] = blo[e]; bf[8 + e] = bhi[e]; }
          c2[t] = __builtin_amdgcn_wmma_f32_16x16x32_f16(
              false, a, false, bf, (short)0, c2[t], false, false);
        }
      }

      // ---- online softmax over j (rows of this batch), diag capture ----
      #pragma unroll
      for (int t = 0; t < 4; ++t) {
        float bb = sb2v[t * 16 + l16];
        float bmax = -3.4e38f;
        #pragma unroll
        for (int r = 0; r < 8; ++r) { c2[t][r] += bb; bmax = fmaxf(bmax, c2[t][r]); }
        bmax = fmaxf(bmax, __shfl_xor(bmax, 16));
        float bsum = 0.f;
        #pragma unroll
        for (int r = 0; r < 8; ++r) bsum += __expf(c2[t][r] - bmax);
        bsum += __shfl_xor(bsum, 16);
        if ((i >> 4) == bj) {                 // diagonal j==i lies in this batch
          int rl = i & 15;
          float dv = c2[t][rl & 7];
          float od = __shfl_xor(dv, 16);
          draw[t] = (half == (rl >> 3)) ? dv : od;
        }
        float nm = fmaxf(rm[t], bmax);
        rs[t] = rs[t] * __expf(rm[t] - nm) + bsum * __expf(bmax - nm);
        rm[t] = nm;
      }
    }  // bj

    // ---- finalize: res = softmax_diag * v ----
    #pragma unroll
    for (int t = 0; t < 4; ++t) {
      int n = t * 16 + l16;
      float res = (__expf(draw[t] - rm[t]) / rs[t]) * sv[i * kD + n];
      if (half == 0) out[b * kM * 128 + i * 128 + n] = res;
    }
  }  // ii
}

extern "C" void kernel_launch(void* const* d_in, const int* in_sizes, int n_in,
                              void* d_out, int out_size, void* d_ws, size_t ws_size,
                              hipStream_t stream) {
  (void)in_sizes; (void)n_in; (void)out_size; (void)d_ws; (void)ws_size;
  const float* pv_xyz   = (const float*)d_in[0];
  const float* features = (const float*)d_in[1];
  const float* Wq  = (const float*)d_in[2];
  const float* bq  = (const float*)d_in[3];
  const float* Wk  = (const float*)d_in[4];
  const float* bk  = (const float*)d_in[5];
  const float* Wv  = (const float*)d_in[6];
  const float* bv  = (const float*)d_in[7];
  const float* pW1 = (const float*)d_in[8];
  const float* pb1 = (const float*)d_in[9];
  const float* plng = (const float*)d_in[10];
  const float* plnb = (const float*)d_in[11];
  const float* pW2 = (const float*)d_in[12];
  const float* pb2 = (const float*)d_in[13];
  const float* ln1g = (const float*)d_in[14];
  const float* ln1b = (const float*)d_in[15];
  const float* wW1 = (const float*)d_in[16];
  const float* wb1 = (const float*)d_in[17];
  const float* ln2g = (const float*)d_in[18];
  const float* ln2b = (const float*)d_in[19];
  const float* wW2 = (const float*)d_in[20];
  const float* wb2 = (const float*)d_in[21];
  float* out = (float*)d_out;

  hipLaunchKernelGGL(point_attn_kernel, dim3(1024), dim3(256), 0, stream,
                     pv_xyz, features, Wq, bq, Wk, bk, Wv, bv,
                     pW1, pb1, plng, plnb, pW2, pb2,
                     ln1g, ln1b, wW1, wb1, ln2g, ln2b, wW2, wb2, out);
}
